// SelfAttentionLayer_17428977287196
// MI455X (gfx1250) — matrix operand order repaired and checked by reading, not compiled, not run on hardware
//
#include <hip/hip_runtime.h>
#include <hip/hip_bf16.h>

#define NT     12544   // B*H*W tokens
#define DMODEL 256
#define DQKV   768
#define DFFN   1024
#define HH     56
#define WWID   56

typedef __attribute__((ext_vector_type(16))) __bf16        v16bf;
typedef __attribute__((ext_vector_type(8)))  float         v8f;
typedef __attribute__((ext_vector_type(8)))  unsigned short u16x8;

struct BFpack { u16x8 lo, hi; };

__device__ __forceinline__ unsigned short f2bfu(float f) {
  unsigned u = __float_as_uint(f);
  return (unsigned short)((u + 0x7FFFu + ((u >> 16) & 1u)) >> 16);  // RNE
}
__device__ __forceinline__ __bf16 f2bf(float f) {
  unsigned short r = f2bfu(f);
  return __builtin_bit_cast(__bf16, r);
}

// ---------------- conversion / weight transpose ----------------
__global__ void cvt_bf16_kernel(const float* __restrict__ in,
                                __bf16* __restrict__ out, int n) {
  int i = blockIdx.x * blockDim.x + threadIdx.x;
  if (i < n) out[i] = f2bf(in[i]);
}

// W is (kd x n) row-major; Wt becomes (n x kd) row-major (K-contiguous rows)
__global__ void transpose_bf16_kernel(const float* __restrict__ w,
                                      __bf16* __restrict__ wt, int kd, int n) {
  int i = blockIdx.x * blockDim.x + threadIdx.x;
  if (i < kd * n) {
    int nn = i / kd, kk = i % kd;
    wt[i] = f2bf(w[(size_t)kk * n + nn]);
  }
}

// ---------------- bf16 WMMA GEMM: C[M,N] = A[M,K] * Bt[N,K]^T + bias ----------------
// One wave -> 32x64 C tile: 8x v_wmma_f32_16x16x32_bf16 per K-step of 32,
// 12 b128 loads per step (B operands reused across two A row-halves).
__global__ void __launch_bounds__(256) gemm_bf16_kernel(
    const __bf16* __restrict__ A, const __bf16* __restrict__ Bt,
    const float* __restrict__ bias, float* __restrict__ Cf,
    __bf16* __restrict__ Cb, int M, int N, int Kd, int relu)
{
  const int lane   = threadIdx.x & 31;
  const int wid    = blockIdx.x * (blockDim.x >> 5) + (threadIdx.x >> 5);
  const int ntile4 = N >> 6;                       // 64-col tiles along N
  if (wid >= (M >> 5) * ntile4) return;            // wave-uniform guard (exact grids)
  const int mt  = wid / ntile4;                    // 32-row tile
  const int nt  = wid % ntile4;
  const int l15 = lane & 15;
  const int hi  = lane >> 4;                       // 0|1

  // A layout: lane<16 holds K=[0..15]; lane>=16 K=[16..31] (matches B)
  const __bf16* Ap0 = A + (size_t)(mt * 32 + l15) * Kd + hi * 16;
  const __bf16* Ap1 = Ap0 + (size_t)16 * Kd;       // rows 16..31 of the tile
  // B layout: lane<16 holds K=[0..15] of col l15; lane>=16 K=[16..31]
  const __bf16* Bp  = Bt + (size_t)(nt * 64 + l15) * Kd + hi * 16;

  v8f c00 = {}, c01 = {}, c02 = {}, c03 = {};
  v8f c10 = {}, c11 = {}, c12 = {}, c13 = {};
  for (int k0 = 0; k0 < Kd; k0 += 32) {
    BFpack ap;
    ap.lo = *(const u16x8*)(Ap0 + k0);
    ap.hi = *(const u16x8*)(Ap0 + k0 + 8);
    v16bf a0 = __builtin_bit_cast(v16bf, ap);
    ap.lo = *(const u16x8*)(Ap1 + k0);
    ap.hi = *(const u16x8*)(Ap1 + k0 + 8);
    v16bf a1 = __builtin_bit_cast(v16bf, ap);
    v16bf b0 = *(const v16bf*)(Bp + k0);
    v16bf b1 = *(const v16bf*)(Bp + (size_t)16 * Kd + k0);
    v16bf b2 = *(const v16bf*)(Bp + (size_t)32 * Kd + k0);
    v16bf b3 = *(const v16bf*)(Bp + (size_t)48 * Kd + k0);
    c00 = __builtin_amdgcn_wmma_f32_16x16x32_bf16(false, a0, false, b0, (short)0, c00, false, false);
    c10 = __builtin_amdgcn_wmma_f32_16x16x32_bf16(false, a1, false, b0, (short)0, c10, false, false);
    c01 = __builtin_amdgcn_wmma_f32_16x16x32_bf16(false, a0, false, b1, (short)0, c01, false, false);
    c11 = __builtin_amdgcn_wmma_f32_16x16x32_bf16(false, a1, false, b1, (short)0, c11, false, false);
    c02 = __builtin_amdgcn_wmma_f32_16x16x32_bf16(false, a0, false, b2, (short)0, c02, false, false);
    c12 = __builtin_amdgcn_wmma_f32_16x16x32_bf16(false, a1, false, b2, (short)0, c12, false, false);
    c03 = __builtin_amdgcn_wmma_f32_16x16x32_bf16(false, a0, false, b3, (short)0, c03, false, false);
    c13 = __builtin_amdgcn_wmma_f32_16x16x32_bf16(false, a1, false, b3, (short)0, c13, false, false);
  }

  // C/D layout: VGPR r -> row tilebase + r + 8*hi, col = tile + l15
  v8f cc[2][4] = { { c00, c01, c02, c03 }, { c10, c11, c12, c13 } };
#pragma unroll
  for (int halfm = 0; halfm < 2; ++halfm) {
    const int rbase = mt * 32 + halfm * 16 + hi * 8;
#pragma unroll
    for (int q = 0; q < 4; ++q) {
      const int col = nt * 64 + q * 16 + l15;
      const float bv = bias[col];
#pragma unroll
      for (int r = 0; r < 8; ++r) {
        float o = cc[halfm][q][r] + bv;
        if (relu) o = fmaxf(o, 0.f);
        size_t idx = (size_t)(rbase + r) * N + col;
        if (Cf) Cf[idx] = o;
        if (Cb) Cb[idx] = f2bf(o);
      }
    }
  }
}

// ---------------- neighborhood attention (online softmax, one wave / token) ----------------
__global__ void __launch_bounds__(256) natt_kernel(const float* __restrict__ qkv,
                                                   __bf16* __restrict__ ctxb)
{
  const int lane = threadIdx.x & 31;
  const int t    = blockIdx.x * (blockDim.x >> 5) + (threadIdx.x >> 5);
  if (t >= NT) return;
  const int w   = t % WWID;
  const int h   = (t / WWID) % HH;
  const int b   = t / (WWID * HH);
  const int hs  = min(max(h - 3, 0), HH - 7);
  const int ws0 = min(max(w - 3, 0), WWID - 7);
  const int head = lane >> 2;
  const int chnk = lane & 3;
  const int doff = head * 32 + chnk * 8;           // 8 dims of one head per lane

  const float4* qp = (const float4*)(qkv + (size_t)t * DQKV + doff);
  float4 qa = qp[0], qb = qp[1];
  const float scl = 0.17677669529663687f;          // 1/sqrt(32)
  qa.x *= scl; qa.y *= scl; qa.z *= scl; qa.w *= scl;
  qb.x *= scl; qb.y *= scl; qb.z *= scl; qb.w *= scl;

  float m = -3.4e38f, l = 0.f;
  float acc[8] = {0.f,0.f,0.f,0.f,0.f,0.f,0.f,0.f};

  for (int j = 0; j < 49; ++j) {
    const int tn = (b * HH + hs + j / 7) * WWID + ws0 + j % 7;
    const float4* kp = (const float4*)(qkv + (size_t)tn * DQKV + 256 + doff);
    float4 k0 = kp[0], k1 = kp[1];
    float s = qa.x*k0.x + qa.y*k0.y + qa.z*k0.z + qa.w*k0.w
            + qb.x*k1.x + qb.y*k1.y + qb.z*k1.z + qb.w*k1.w;
    s += __shfl_xor(s, 1, 32);                     // reduce over the 4-lane head group
    s += __shfl_xor(s, 2, 32);
    const float mn = fmaxf(m, s);
    const float sc = __expf(m - mn);
    const float p  = __expf(s - mn);
    l = l * sc + p;
    const float4* vp = (const float4*)(qkv + (size_t)tn * DQKV + 512 + doff);
    float4 v0 = vp[0], v1 = vp[1];
    acc[0] = acc[0]*sc + p*v0.x;  acc[1] = acc[1]*sc + p*v0.y;
    acc[2] = acc[2]*sc + p*v0.z;  acc[3] = acc[3]*sc + p*v0.w;
    acc[4] = acc[4]*sc + p*v1.x;  acc[5] = acc[5]*sc + p*v1.y;
    acc[6] = acc[6]*sc + p*v1.z;  acc[7] = acc[7]*sc + p*v1.w;
    m = mn;
  }
  const float inv = 1.f / l;
  u16x8 ov;
#pragma unroll
  for (int i = 0; i < 8; ++i) ov[i] = f2bfu(acc[i] * inv);
  *(u16x8*)(ctxb + (size_t)t * DMODEL + doff) = ov;
}

// ---------------- residual + LayerNorm (one wave / token) ----------------
__global__ void __launch_bounds__(256) ln_kernel(const float* __restrict__ xa,
                                                 const float* __restrict__ xb,
                                                 const float* __restrict__ g,
                                                 const float* __restrict__ be,
                                                 float* __restrict__ outf,
                                                 __bf16* __restrict__ outb, int ntok)
{
  const int lane = threadIdx.x & 31;
  const int t    = blockIdx.x * (blockDim.x >> 5) + (threadIdx.x >> 5);
  if (t >= ntok) return;
  const float4* pa = (const float4*)(xa + (size_t)t * DMODEL + lane * 8);
  const float4* pb = (const float4*)(xb + (size_t)t * DMODEL + lane * 8);
  float v[8];
  float4 a0 = pa[0], a1 = pa[1], b0 = pb[0], b1 = pb[1];
  v[0]=a0.x+b0.x; v[1]=a0.y+b0.y; v[2]=a0.z+b0.z; v[3]=a0.w+b0.w;
  v[4]=a1.x+b1.x; v[5]=a1.y+b1.y; v[6]=a1.z+b1.z; v[7]=a1.w+b1.w;
  float s = 0.f;
#pragma unroll
  for (int i = 0; i < 8; ++i) s += v[i];
#pragma unroll
  for (int msk = 1; msk < 32; msk <<= 1) s += __shfl_xor(s, msk, 32);
  const float mean = s * (1.f / DMODEL);
  float var = 0.f;
#pragma unroll
  for (int i = 0; i < 8; ++i) { float d = v[i] - mean; var += d * d; }
#pragma unroll
  for (int msk = 1; msk < 32; msk <<= 1) var += __shfl_xor(var, msk, 32);
  const float rstd = rsqrtf(var * (1.f / DMODEL) + 1e-5f);
#pragma unroll
  for (int i = 0; i < 8; ++i) {
    const int col = lane * 8 + i;
    const float o = (v[i] - mean) * rstd * g[col] + be[col];
    const size_t idx = (size_t)t * DMODEL + col;
    if (outf) outf[idx] = o;
    if (outb) outb[idx] = f2bf(o);
  }
}

// ---------------- launcher ----------------
extern "C" void kernel_launch(void* const* d_in, const int* in_sizes, int n_in,
                              void* d_out, int out_size, void* d_ws, size_t ws_size,
                              hipStream_t stream)
{
  const float* x     = (const float*)d_in[0];
  const float* Wqkv  = (const float*)d_in[1];
  const float* bqkv  = (const float*)d_in[2];
  const float* Wproj = (const float*)d_in[3];
  const float* bproj = (const float*)d_in[4];
  const float* W1    = (const float*)d_in[5];
  const float* b1    = (const float*)d_in[6];
  const float* W2    = (const float*)d_in[7];
  const float* b2    = (const float*)d_in[8];
  const float* g1    = (const float*)d_in[9];
  const float* be1   = (const float*)d_in[10];
  const float* g2    = (const float*)d_in[11];
  const float* be2   = (const float*)d_in[12];
  float* out = (float*)d_out;

  char* wsp = (char*)d_ws;
  size_t off = 0;
  auto carve = [&](size_t bytes) -> void* {
    void* p = wsp + off;
    off += (bytes + 255) & ~(size_t)255;
    return p;
  };
  __bf16* xb     = (__bf16*)carve((size_t)NT * DMODEL * 2);
  __bf16* Wqkvt  = (__bf16*)carve((size_t)DQKV  * DMODEL * 2);
  __bf16* Wprojt = (__bf16*)carve((size_t)DMODEL * DMODEL * 2);
  __bf16* W1t    = (__bf16*)carve((size_t)DFFN  * DMODEL * 2);
  __bf16* W2t    = (__bf16*)carve((size_t)DMODEL * DFFN  * 2);
  void*   big    = carve((size_t)NT * DQKV * 4);      // qkv f32; reused as h bf16
  float*  qkv    = (float*)big;
  __bf16* hb     = (__bf16*)big;
  __bf16* ctxb   = (__bf16*)carve((size_t)NT * DMODEL * 2);
  void*   mid    = carve((size_t)NT * DMODEL * 4);    // proj out f32; reused as ffn out f32
  float*  y1     = (float*)mid;
  float*  ff     = (float*)mid;
  float*  x1f    = (float*)carve((size_t)NT * DMODEL * 4);
  __bf16* x1b    = (__bf16*)carve((size_t)NT * DMODEL * 2);

  // 1) x -> bf16
  { int n = NT * DMODEL; cvt_bf16_kernel<<<(n + 255) / 256, 256, 0, stream>>>(x, xb, n); }
  // 2) transpose weights -> bf16 (N x K row-major)
  { int n = DMODEL * DQKV;  transpose_bf16_kernel<<<(n + 255) / 256, 256, 0, stream>>>(Wqkv,  Wqkvt,  DMODEL, DQKV); }
  { int n = DMODEL * DMODEL;transpose_bf16_kernel<<<(n + 255) / 256, 256, 0, stream>>>(Wproj, Wprojt, DMODEL, DMODEL); }
  { int n = DMODEL * DFFN;  transpose_bf16_kernel<<<(n + 255) / 256, 256, 0, stream>>>(W1,    W1t,    DMODEL, DFFN); }
  { int n = DFFN * DMODEL;  transpose_bf16_kernel<<<(n + 255) / 256, 256, 0, stream>>>(W2,    W2t,    DFFN,   DMODEL); }

  // 3) qkv = x @ Wqkv + b   (12544x768, K=256) -> f32
  { int waves = (NT >> 5) * (DQKV >> 6);
    gemm_bf16_kernel<<<waves / 8, 256, 0, stream>>>(xb, Wqkvt, bqkv, qkv, nullptr, NT, DQKV, DMODEL, 0); }
  // 4) neighborhood attention -> ctx bf16
  natt_kernel<<<NT / 8, 256, 0, stream>>>(qkv, ctxb);
  // 5) y1 = ctx @ Wproj + b  (12544x256, K=256) -> f32
  { int waves = (NT >> 5) * (DMODEL >> 6);
    gemm_bf16_kernel<<<waves / 8, 256, 0, stream>>>(ctxb, Wprojt, bproj, y1, nullptr, NT, DMODEL, DMODEL, 0); }
  // 6) x1 = LN(x + y1) -> f32 + bf16
  ln_kernel<<<NT / 8, 256, 0, stream>>>(x, y1, g1, be1, x1f, x1b, NT);
  // 7) h = relu(x1 @ W1 + b1) (12544x1024, K=256) -> bf16 (reuses qkv region)
  { int waves = (NT >> 5) * (DFFN >> 6);
    gemm_bf16_kernel<<<waves / 8, 256, 0, stream>>>(x1b, W1t, b1, nullptr, hb, NT, DFFN, DMODEL, 1); }
  // 8) f = h @ W2 + b2 (12544x256, K=1024) -> f32 (reuses y1 region)
  { int waves = (NT >> 5) * (DMODEL >> 6);
    gemm_bf16_kernel<<<waves / 8, 256, 0, stream>>>(hb, W2t, b2, ff, nullptr, NT, DMODEL, DFFN, 0); }
  // 9) out = LN(x1 + f)
  ln_kernel<<<NT / 8, 256, 0, stream>>>(x1f, ff, g2, be2, out, nullptr, NT);
}
